// HanDecoder_42743514530128
// MI455X (gfx1250) — compile-verified, matrix operand order checked by reading, not run on hardware
//
#include <hip/hip_runtime.h>

#define NOTES 8192
#define NPB 8
#define BEATS 1024
#define MEAS 256
#define NOTE_D 512
#define BEAT_D 512
#define MEAS_D 256
#define ENC 128
#define TEMPO_H 256
#define FINAL_H 256
#define NOUT 11
#define KN 1408          // static input dim for output LSTM (1419 - 11 dynamic)
#define KT 928           // static input dim for tempo LSTM, 904 padded to 928
#define GN 1024          // 4*FINAL_H gates
#define GT 1024          // 4*TEMPO_H gates

typedef __bf16 bf16;
typedef __attribute__((ext_vector_type(16))) __bf16 v16bf;
typedef __attribute__((ext_vector_type(8)))  float  v8f;

__device__ __forceinline__ float sigmoidf_(float x){ return 1.0f/(1.0f+__expf(-x)); }

// ---------------- measure LSTM (256 sequential steps) + style expandor + mpf fc
__global__ __launch_bounds__(512)
void psm_kernel(const float* __restrict__ perf, const float* __restrict__ sveW,
                const float* __restrict__ sveB, const float* __restrict__ meas,
                const float* __restrict__ Wih, const float* __restrict__ Whh,
                const float* __restrict__ bias, const float* __restrict__ mpfW,
                const float* __restrict__ mpfB, float* __restrict__ mps)
{
    __shared__ float x[ENC + MEAS_D];   // [z(128), measure_emb(256)]
    __shared__ float h[ENC];
    __shared__ float zg[4*ENC];
    const int t = threadIdx.x;
    float c = 0.f;
    if (t < ENC) {
        float s = sveB[t];
        for (int j = 0; j < 64; ++j) s += sveW[t*64 + j] * perf[j];
        x[t] = fmaxf(s, 0.f);           // relu(style vector)
        h[t] = 0.f;
    }
    __syncthreads();
    for (int m = 0; m < MEAS; ++m) {
        if (t < MEAS_D) x[ENC + t] = meas[(size_t)m*MEAS_D + t];
        __syncthreads();
        {
            float s = bias[t];
            const float* wr = Wih + (size_t)t*(ENC+MEAS_D);
            for (int j = 0; j < ENC+MEAS_D; ++j) s += wr[j]*x[j];
            const float* hr = Whh + (size_t)t*ENC;
            for (int j = 0; j < ENC; ++j) s += hr[j]*h[j];
            zg[t] = s;
        }
        __syncthreads();
        if (t < ENC) {
            float ig = sigmoidf_(zg[t]);
            float fg = sigmoidf_(zg[ENC+t]);
            float gg = tanhf(zg[2*ENC+t]);
            float og = sigmoidf_(zg[3*ENC+t]);
            c = fg*c + ig*gg;
            h[t] = og*tanhf(c);
        }
        __syncthreads();
        if (t < ENC) {
            float s = mpfB[t];
            for (int j = 0; j < ENC; ++j) s += mpfW[t*ENC + j]*h[j];
            mps[(size_t)m*ENC + t] = s;  // measure_perform_style
        }
        __syncthreads();
    }
}

// ---------------- pack static feature matrices / weights to bf16
__global__ void pack_note_A(const float* __restrict__ ne, const float* __restrict__ be,
                            const float* __restrict__ me, const float* __restrict__ mps,
                            bf16* __restrict__ A)
{
    size_t idx = (size_t)blockIdx.x*blockDim.x + threadIdx.x;
    if (idx >= (size_t)NOTES*KN) return;
    int note = (int)(idx / KN), j = (int)(idx % KN);
    float v;
    if (j < NOTE_D)              v = ne[(size_t)note*NOTE_D + j];
    else if (j < NOTE_D+BEAT_D)  v = be[(size_t)(note>>3)*BEAT_D + (j-NOTE_D)];
    else if (j < 1280)           v = me[(size_t)(note>>5)*MEAS_D + (j-1024)];
    else                         v = mps[(size_t)(note>>5)*ENC + (j-1280)];
    A[idx] = (bf16)v;
}

__global__ void pack_note_W(const float* __restrict__ W, bf16* __restrict__ o)
{
    size_t idx = (size_t)blockIdx.x*blockDim.x + threadIdx.x;
    if (idx >= (size_t)GN*KN) return;
    int g = (int)(idx / KN), j = (int)(idx % KN);
    int src = (j < 1280) ? j : j + 11;   // skip 11 dynamic prev_out columns
    o[idx] = (bf16)W[(size_t)g*1419 + src];
}

__global__ void pack_tempo_A(const float* __restrict__ be, const float* __restrict__ me,
                             const float* __restrict__ res, const float* __restrict__ mps,
                             bf16* __restrict__ A)
{
    size_t idx = (size_t)blockIdx.x*blockDim.x + threadIdx.x;
    if (idx >= (size_t)BEATS*KT) return;
    int beat = (int)(idx / KT), j = (int)(idx % KT);
    int mb = beat >> 2;
    float v = 0.f;
    if (j < BEAT_D)      v = be[(size_t)beat*BEAT_D + j];
    else if (j < 768)    v = me[(size_t)mb*MEAS_D + (j-512)];
    else if (j < 776)    v = res[(size_t)beat*8 + (j-768)];
    else if (j < 904)    v = mps[(size_t)mb*ENC + (j-776)];
    A[idx] = (bf16)v;    // j in [904,928): zero pad
}

__global__ void pack_tempo_W(const float* __restrict__ W, bf16* __restrict__ o)
{
    size_t idx = (size_t)blockIdx.x*blockDim.x + threadIdx.x;
    if (idx >= (size_t)GT*KT) return;
    int g = (int)(idx / KT), j = (int)(idx % KT);
    float v = 0.f;
    int src = -1;
    if (j < 768)       src = j;        // beat_emb + measure_emb
    else if (j < 776)  src = j + 1;    // res_info (skip prev_tempo col 768)
    else if (j < 904)  src = j + 11;   // mps (skip result cols 777..786)
    if (src >= 0) v = W[(size_t)g*915 + src];
    o[idx] = (bf16)v;
}

// ---------------- bf16 WMMA GEMM: C[M,N] = A[M,K] @ W[N,K]^T + bias[N]
// One wave per 32x32 output tile: 2 A-frags x 2 W-frags -> 4 accumulators, so each
// 128-bit load feeds half a v_wmma (8 loads : 4 WMMAs per K-step) instead of
// the 4:1 ratio of a 16x16 tile. Fragments follow the 16-bit A/B VGPR layout
// (lane&15 -> row/col, lane>>4 selects K sub-halves {0..7,16..23} vs {8..15,24..31}).
__global__ __launch_bounds__(256)
void wmma_gemm(const bf16* __restrict__ A, const bf16* __restrict__ W,
               const float* __restrict__ bias, float* __restrict__ C,
               int M, int N, int K)
{
    int wave = (int)((blockIdx.x * blockDim.x + threadIdx.x) >> 5);
    int lane = threadIdx.x & 31;
    int tilesN = N >> 5;                          // 32-wide N tiles
    int mT = wave / tilesN;
    int nT = wave - mT*tilesN;
    if (mT*32 >= M) return;                       // wave-uniform guard: EXEC stays all-1s
    int r = lane & 15;
    int khalf = (lane >> 4) << 3;
    const bf16* a0 = A + (size_t)(mT*32 + r)      * K + khalf;
    const bf16* a1 = A + (size_t)(mT*32 + 16 + r) * K + khalf;
    const bf16* w0 = W + (size_t)(nT*32 + r)      * K + khalf;
    const bf16* w1 = W + (size_t)(nT*32 + 16 + r) * K + khalf;
    float b0 = bias[nT*32 + r];
    float b1 = bias[nT*32 + 16 + r];
    v8f acc00, acc01, acc10, acc11;
#pragma unroll
    for (int i = 0; i < 8; ++i) { acc00[i] = b0; acc01[i] = b1; acc10[i] = b0; acc11[i] = b1; }
    union Frag { v16bf v; uint4 u[2]; };
    for (int kk = 0; kk < K; kk += 32) {
        Frag fa0, fa1, fw0, fw1;
        fa0.u[0] = *(const uint4*)(a0 + kk);
        fa0.u[1] = *(const uint4*)(a0 + kk + 16);
        fa1.u[0] = *(const uint4*)(a1 + kk);
        fa1.u[1] = *(const uint4*)(a1 + kk + 16);
        fw0.u[0] = *(const uint4*)(w0 + kk);
        fw0.u[1] = *(const uint4*)(w0 + kk + 16);
        fw1.u[0] = *(const uint4*)(w1 + kk);
        fw1.u[1] = *(const uint4*)(w1 + kk + 16);
        // cover L2 latency for the next K block (lowers to global_prefetch_b8)
        __builtin_prefetch(a0 + kk + 32, 0, 1);
        __builtin_prefetch(a1 + kk + 32, 0, 1);
        acc00 = __builtin_amdgcn_wmma_f32_16x16x32_bf16(false, fa0.v, false, fw0.v,
                                                        (short)0, acc00, false, false);
        acc01 = __builtin_amdgcn_wmma_f32_16x16x32_bf16(false, fa0.v, false, fw1.v,
                                                        (short)0, acc01, false, false);
        acc10 = __builtin_amdgcn_wmma_f32_16x16x32_bf16(false, fa1.v, false, fw0.v,
                                                        (short)0, acc10, false, false);
        acc11 = __builtin_amdgcn_wmma_f32_16x16x32_bf16(false, fa1.v, false, fw1.v,
                                                        (short)0, acc11, false, false);
    }
    int mBase = mT*32 + ((lane >> 4) << 3);       // vgpr i -> M = i + 8*(lane>>4)
    int n0 = nT*32 + r, n1 = n0 + 16;
#pragma unroll
    for (int i = 0; i < 8; ++i) {
        C[(size_t)(mBase + i)*N + n0]      = acc00[i];
        C[(size_t)(mBase + i)*N + n1]      = acc01[i];
        C[(size_t)(mBase + 16 + i)*N + n0] = acc10[i];
        C[(size_t)(mBase + 16 + i)*N + n1] = acc11[i];
    }
}

// ---------------- persistent sequential scan over 8192 notes (1 workgroup, 32 waves)
__global__ __launch_bounds__(1024)
void scan_kernel(const float* __restrict__ Zs, const float* __restrict__ Zt,
                 const float* __restrict__ outWih, const float* __restrict__ outWhh,
                 const float* __restrict__ tempoWih, const float* __restrict__ tempoWhh,
                 const float* __restrict__ fcW, const float* __restrict__ fcB,
                 const float* __restrict__ attW, const float* __restrict__ attB,
                 const float* __restrict__ attC,
                 const float* __restrict__ tfcW, const float* __restrict__ tfcB,
                 float* __restrict__ out)
{
    __shared__ float h[FINAL_H];
    __shared__ float th[TEMPO_H];
    __shared__ float zg[1024];
    __shared__ float s_fcW[10*FINAL_H];
    __shared__ float s_fcB[10];
    __shared__ float s_attW[100], s_attB[10], s_attC[10];
    __shared__ float s_tfcW[TEMPO_H];
    __shared__ float buf[NPB*10];
    __shared__ float prevout[NOUT];
    __shared__ float result[10];
    __shared__ float sc[8];
    __shared__ float pt;                // prev_tempo
    const int t = threadIdx.x;          // gate index 0..1023

    // per-thread dynamic-column weights (the truly recurrent inputs)
    float wdo[11];
#pragma unroll
    for (int j = 0; j < 11; ++j) wdo[j] = outWih[(size_t)t*1419 + 1280 + j];
    float wdt[11];
    wdt[0] = tempoWih[(size_t)t*915 + 768];        // prev_tempo column
#pragma unroll
    for (int j = 0; j < 10; ++j) wdt[1+j] = tempoWih[(size_t)t*915 + 777 + j]; // result cols

    for (int i = t; i < 10*FINAL_H; i += 1024) s_fcW[i] = fcW[i];
    if (t < 10)  { s_fcB[t] = fcB[t]; s_attB[t] = attB[t]; s_attC[t] = attC[t]; }
    if (t < 100)   s_attW[t] = attW[t];
    if (t < TEMPO_H) { s_tfcW[t] = tfcW[t]; th[t] = 0.f; h[t] = 0.f; }
    if (t < NPB*10)  buf[t] = 0.f;
    if (t < NOUT)    prevout[t] = 0.f;
    if (t < 10)      result[t] = 0.f;
    if (t == 0)      pt = 0.f;
    float fcc = 0.f, tcc = 0.f;
    __syncthreads();

    for (int note = 0; note < NOTES; ++note) {
        if ((note & (NPB-1)) == 0) {
            const int beat = note >> 3;
            // attention over previous beat's 8 fc outputs
            if (t < 8) {
                float s_i = 0.f;
                for (int j = 0; j < 10; ++j) {
                    float a = s_attB[j];
                    for (int k = 0; k < 10; ++k) a += buf[t*10+k]*s_attW[j*10+k];
                    s_i += tanhf(a)*s_attC[j];
                }
                sc[t] = s_i;
            }
            __syncthreads();
            if (t == 0) {
                if (note == 0) {
                    for (int k = 0; k < 10; ++k) result[k] = 0.f;
                } else {
                    float mx = sc[0];
                    for (int i = 1; i < 8; ++i) mx = fmaxf(mx, sc[i]);
                    float w[8]; float den = 0.f;
                    for (int i = 0; i < 8; ++i) { w[i] = __expf(sc[i]-mx); den += w[i]; }
                    float inv = 1.f/den;
                    for (int k = 0; k < 10; ++k) {
                        float r = 0.f;
                        for (int i = 0; i < 8; ++i) r += w[i]*buf[i*10+k];
                        result[k] = r*inv;
                    }
                }
            }
            __syncthreads();
            // tempo LSTM gate t: static(+bias) precomputed in Zt by WMMA GEMM
            float s = Zt[(size_t)beat*GT + t] + wdt[0]*pt;
#pragma unroll
            for (int j = 0; j < 10; ++j) s += wdt[1+j]*result[j];
            const float4* wr = (const float4*)(tempoWhh + (size_t)t*TEMPO_H);
            for (int j = 0; j < TEMPO_H/4; ++j) {
                float4 w4 = wr[j];
                s += w4.x*th[4*j] + w4.y*th[4*j+1] + w4.z*th[4*j+2] + w4.w*th[4*j+3];
            }
            zg[t] = s;
            __syncthreads();
            if (t < TEMPO_H) {
                float ig = sigmoidf_(zg[t]);
                float fg = sigmoidf_(zg[TEMPO_H+t]);
                float gg = tanhf(zg[2*TEMPO_H+t]);
                float og = sigmoidf_(zg[3*TEMPO_H+t]);
                tcc = fg*tcc + ig*gg;
                th[t] = og*tanhf(tcc);
            }
            __syncthreads();
            if (t == 0) {
                float p = tfcB[0];
                for (int j = 0; j < TEMPO_H; ++j) p += s_tfcW[j]*th[j];
                pt = p;
            }
            __syncthreads();
        }
        // output LSTM gate t: static(+bias) precomputed in Zs by WMMA GEMM
        float s = Zs[(size_t)note*GN + t];
#pragma unroll
        for (int j = 0; j < 11; ++j) s += wdo[j]*prevout[j];
        const float4* wr = (const float4*)(outWhh + (size_t)t*FINAL_H);
        for (int j = 0; j < FINAL_H/4; ++j) {
            float4 w4 = wr[j];
            s += w4.x*h[4*j] + w4.y*h[4*j+1] + w4.z*h[4*j+2] + w4.w*h[4*j+3];
        }
        zg[t] = s;
        __syncthreads();
        if (t < FINAL_H) {
            float ig = sigmoidf_(zg[t]);
            float fg = sigmoidf_(zg[FINAL_H+t]);
            float gg = tanhf(zg[2*FINAL_H+t]);
            float og = sigmoidf_(zg[3*FINAL_H+t]);
            fcc = fg*fcc + ig*gg;
            h[t] = og*tanhf(fcc);
        }
        __syncthreads();
        if (t < 10) {
            float o = s_fcB[t];
            for (int j = 0; j < FINAL_H; ++j) o += s_fcW[t*FINAL_H+j]*h[j];
            buf[(note & 7)*10 + t] = o;
            prevout[1+t] = o;
            out[(size_t)note*NOUT + 1 + t] = o;
        }
        if (t == 10) {
            prevout[0] = pt;
            out[(size_t)note*NOUT] = pt;
        }
        __syncthreads();
    }
}

extern "C" void kernel_launch(void* const* d_in, const int* in_sizes, int n_in,
                              void* d_out, int out_size, void* d_ws, size_t ws_size,
                              hipStream_t stream)
{
    (void)in_sizes; (void)n_in; (void)out_size; (void)ws_size;
    const float* note_emb    = (const float*)d_in[0];
    const float* beat_emb    = (const float*)d_in[1];
    const float* measure_emb = (const float*)d_in[2];
    const float* perf        = (const float*)d_in[3];
    const float* res_info    = (const float*)d_in[4];
    const float* sve_W       = (const float*)d_in[5];
    const float* sve_b       = (const float*)d_in[6];
    const float* psm_Wih     = (const float*)d_in[7];
    const float* psm_Whh     = (const float*)d_in[8];
    const float* psm_b       = (const float*)d_in[9];
    const float* mpf_W       = (const float*)d_in[10];
    const float* mpf_b       = (const float*)d_in[11];
    const float* att_W       = (const float*)d_in[12];
    const float* att_b       = (const float*)d_in[13];
    const float* att_c       = (const float*)d_in[14];
    const float* tempo_Wih   = (const float*)d_in[15];
    const float* tempo_Whh   = (const float*)d_in[16];
    const float* tempo_b     = (const float*)d_in[17];
    const float* tempo_fc_W  = (const float*)d_in[18];
    const float* tempo_fc_b  = (const float*)d_in[19];
    const float* out_Wih     = (const float*)d_in[20];
    const float* out_Whh     = (const float*)d_in[21];
    const float* out_b       = (const float*)d_in[22];
    const float* fc_W        = (const float*)d_in[23];
    const float* fc_b        = (const float*)d_in[24];
    float* out = (float*)d_out;

    char* ws = (char*)d_ws;
    float* mps = (float*)(ws + 0);          //   256*128*4   = 131072
    float* Zs  = (float*)(ws + 131072);     //  8192*1024*4  = 33554432
    float* Zt  = (float*)(ws + 33685504);   //  1024*1024*4  = 4194304
    bf16*  An  = (bf16*) (ws + 37879808);   //  8192*1408*2  = 23068672
    bf16*  Wn  = (bf16*) (ws + 60948480);   //  1024*1408*2  = 2883584
    bf16*  At  = (bf16*) (ws + 63832064);   //  1024*928*2   = 1900544
    bf16*  Wt  = (bf16*) (ws + 65732608);   //  1024*928*2   = 1900544

    psm_kernel<<<1, 512, 0, stream>>>(perf, sve_W, sve_b, measure_emb,
                                      psm_Wih, psm_Whh, psm_b, mpf_W, mpf_b, mps);

    {
        size_t n = (size_t)NOTES*KN;
        pack_note_A<<<(unsigned)((n + 255)/256), 256, 0, stream>>>(note_emb, beat_emb,
                                                                   measure_emb, mps, An);
    }
    {
        size_t n = (size_t)GN*KN;
        pack_note_W<<<(unsigned)((n + 255)/256), 256, 0, stream>>>(out_Wih, Wn);
    }
    {
        size_t n = (size_t)BEATS*KT;
        pack_tempo_A<<<(unsigned)((n + 255)/256), 256, 0, stream>>>(beat_emb, measure_emb,
                                                                    res_info, mps, At);
        pack_tempo_W<<<(unsigned)((n + 255)/256), 256, 0, stream>>>(tempo_Wih, Wt);
    }
    {
        int waves = (NOTES/32)*(GN/32);     // 8192 32x32 tiles, 8 waves/block
        wmma_gemm<<<waves/8, 256, 0, stream>>>(An, Wn, out_b, Zs, NOTES, GN, KN);
    }
    {
        int waves = (BEATS/32)*(GT/32);     // 1024 32x32 tiles
        wmma_gemm<<<waves/8, 256, 0, stream>>>(At, Wt, tempo_b, Zt, BEATS, GT, KT);
    }
    scan_kernel<<<1, 1024, 0, stream>>>(Zs, Zt, out_Wih, out_Whh, tempo_Wih, tempo_Whh,
                                        fc_W, fc_b, att_W, att_b, att_c,
                                        tempo_fc_W, tempo_fc_b, out);
}
